// AMDOptimizedAttention_7378753815250
// MI455X (gfx1250) — matrix operand order repaired and checked by reading, not compiled, not run on hardware
//
#include <hip/hip_runtime.h>
#include <hip/hip_bf16.h>

// ---------------------------------------------------------------------------
// MI455X / gfx1250 attention layer: bf16 WMMA everywhere, fp32 softmax stats,
// async global->LDS staging (ASYNCcnt) where layout permits.
// B=2, S=2048, H=2048, NH=16, HD=128.
// ---------------------------------------------------------------------------

typedef __bf16 bf16_t;
typedef __attribute__((ext_vector_type(8)))  __bf16 v8bf;
typedef __attribute__((ext_vector_type(16))) __bf16 v16bf;
typedef __attribute__((ext_vector_type(8)))  float  v8f;

#define BATCH 2
#define SEQ   2048
#define HID   2048
#define NHEAD 16
#define HDIM  128
#define MTOT  (BATCH * SEQ)   // 4096

#if defined(__has_builtin)
#  if __has_builtin(__builtin_amdgcn_global_load_async_to_lds_b128) && \
      __has_builtin(__builtin_amdgcn_s_wait_asynccnt)
#    define HAVE_ASYNC_LDS 1
#  endif
#endif

#ifdef HAVE_ASYNC_LDS
typedef int v4i __attribute__((vector_size(4 * sizeof(int))));
typedef __attribute__((address_space(1))) v4i gv4i;   // global (AS1) int4
typedef __attribute__((address_space(3))) v4i lv4i;   // LDS (AS3) int4
#endif

__device__ __forceinline__ float bf2f(bf16_t x) { return (float)x; }
__device__ __forceinline__ bf16_t f2bf(float x) { return (bf16_t)x; }

// 16-byte global -> LDS copy; async DMA (ASYNCcnt) when the toolchain has it.
__device__ __forceinline__ void lds_copy16(const bf16_t* g, bf16_t* l) {
#ifdef HAVE_ASYNC_LDS
  __builtin_amdgcn_global_load_async_to_lds_b128((gv4i*)g, (lv4i*)l, 0, 0);
#else
  *(uint4*)l = *(const uint4*)g;
#endif
}
__device__ __forceinline__ void async_wait0() {
#ifdef HAVE_ASYNC_LDS
  __builtin_amdgcn_s_wait_asynccnt(0);
#endif
}

__device__ __forceinline__ v16bf cat16(v8bf lo, v8bf hi) {
  return __builtin_shufflevector(lo, hi, 0,1,2,3,4,5,6,7,8,9,10,11,12,13,14,15);
}

__device__ __forceinline__ float half16_max(float v) {
  for (int m = 1; m < 16; m <<= 1) v = fmaxf(v, __shfl_xor(v, m, 32));
  return v;
}
__device__ __forceinline__ float half16_sum(float v) {
  for (int m = 1; m < 16; m <<= 1) v += __shfl_xor(v, m, 32);
  return v;
}

// ---------------------------------------------------------------------------
// Stage 1: fp32 -> bf16 conversion (bandwidth pass)
// ---------------------------------------------------------------------------
__global__ void cvt_f32_bf16(const float* __restrict__ in,
                             bf16_t* __restrict__ out, int n) {
  int i = (blockIdx.x * blockDim.x + threadIdx.x) * 4;
  if (i < n) {
    float4 v = *(const float4*)(in + i);
    out[i + 0] = f2bf(v.x);
    out[i + 1] = f2bf(v.y);
    out[i + 2] = f2bf(v.z);
    out[i + 3] = f2bf(v.w);
  }
}

// ---------------------------------------------------------------------------
// Stage 2/5: Y[M,N] = A[M,K] @ W[N,K]^T, bf16 WMMA, f32 acc.
// 256 thr (8 waves), 128x128 tile, K-step 32, double-buffered LDS with
// async global->LDS for the A stream.
// ---------------------------------------------------------------------------
__global__ __launch_bounds__(256)
void gemm_xwt(const bf16_t* __restrict__ A, const bf16_t* __restrict__ W,
              float* __restrict__ outF, bf16_t* __restrict__ outB,
              int Mdim, int Ndim, int Kdim) {
  __shared__ bf16_t As[2][128 * 32];  // [m][k]
  __shared__ bf16_t Bs[2][32 * 128];  // [k][n]  (transposed W tile)

  const int lane = threadIdx.x & 31;
  const int w    = threadIdx.x >> 5;
  const int wm   = w & 3;   // 0..3 -> 32-row strip
  const int wn   = w >> 2;  // 0..1 -> 64-col strip
  const int m0   = blockIdx.y * 128;
  const int n0   = blockIdx.x * 128;

  v8f acc[2][4] = {};

  const int lrow = threadIdx.x >> 1;        // 0..127
  const int lkh  = (threadIdx.x & 1) * 16;  // 0 or 16
  const int kb8  = (lane >> 4) * 8;
  const int nTiles = Kdim >> 5;

  const bf16_t* aSrc = &A[(size_t)(m0 + lrow) * Kdim + lkh];
  const bf16_t* wSrc = &W[(size_t)(n0 + lrow) * Kdim + lkh];

  // ---- prologue: stage K-tile 0 into buffer 0 ----
  lds_copy16(aSrc, &As[0][lrow * 32 + lkh]);
  {
    union { uint4 u; bf16_t h[16]; } ub;
    ub.u = *(const uint4*)wSrc;
#pragma unroll
    for (int e = 0; e < 16; ++e) Bs[0][(lkh + e) * 128 + lrow] = ub.h[e];
  }
  async_wait0();
  __syncthreads();

  for (int i = 0; i < nTiles; ++i) {
    const int cur = i & 1;
    const int nxt = cur ^ 1;

    // ---- kick off next tile's staging (overlaps with WMMA below) ----
    if (i + 1 < nTiles) {
      const bf16_t* an  = aSrc + (size_t)(i + 1) * 32;
      const bf16_t* wn_ = wSrc + (size_t)(i + 1) * 32;
      lds_copy16(an, &As[nxt][lrow * 32 + lkh]);
      union { uint4 u; bf16_t h[16]; } ub;
      ub.u = *(const uint4*)wn_;
#pragma unroll
      for (int e = 0; e < 16; ++e) Bs[nxt][(lkh + e) * 128 + lrow] = ub.h[e];
      if (i + 2 < nTiles) {
        __builtin_prefetch(aSrc + (size_t)(i + 2) * 32, 0, 3);
        __builtin_prefetch(wSrc + (size_t)(i + 2) * 32, 0, 3);
      }
    }

    // ---- compute on current buffer ----
    v16bf af[2], bfr[4];
#pragma unroll
    for (int mb = 0; mb < 2; ++mb) {
      const bf16_t* ap = &As[cur][(wm * 32 + mb * 16 + (lane & 15)) * 32 + kb8];
      af[mb] = cat16(*(const v8bf*)ap, *(const v8bf*)(ap + 16));
    }
#pragma unroll
    for (int nb = 0; nb < 4; ++nb) {
      for (int s = 0; s < 16; ++s)
        bfr[nb][s] = Bs[cur][(kb8 + (s & 7) + (s >> 3) * 16) * 128 + wn * 64 + nb * 16 + (lane & 15)];
    }
#pragma unroll
    for (int mb = 0; mb < 2; ++mb)
#pragma unroll
      for (int nb = 0; nb < 4; ++nb)
        acc[mb][nb] = __builtin_amdgcn_wmma_f32_16x16x32_bf16(
            false, af[mb], false, bfr[nb], (short)0, acc[mb][nb], false, false);

    async_wait0();
    __syncthreads();
  }

#pragma unroll
  for (int mb = 0; mb < 2; ++mb)
#pragma unroll
    for (int nb = 0; nb < 4; ++nb)
#pragma unroll
      for (int r = 0; r < 8; ++r) {
        int gm = m0 + wm * 32 + mb * 16 + (lane >> 4) * 8 + r;
        int gn = n0 + wn * 64 + nb * 16 + (lane & 15);
        if (outF) outF[(size_t)gm * Ndim + gn] = acc[mb][nb][r];
        else      outB[(size_t)gm * Ndim + gn] = f2bf(acc[mb][nb][r]);
      }
}

// ---------------------------------------------------------------------------
// Stage 3: RoPE + [B,S,NH,HD] -> [B,NH,S,HD] transpose (bf16)
// ---------------------------------------------------------------------------
__global__ void rope_transpose(const bf16_t* __restrict__ src,
                               bf16_t* __restrict__ dst, int applyRope) {
  int idx  = blockIdx.x * blockDim.x + threadIdx.x;  // (b,h,s,half)
  int half = idx & 63;
  int t    = idx >> 6;
  int s    = t & (SEQ - 1);  t >>= 11;
  int h    = t & (NHEAD - 1);
  int b    = t >> 4;

  size_t srcRow  = ((size_t)b * SEQ + s) * HID + (size_t)h * HDIM;
  size_t dstBase = (((size_t)b * NHEAD + h) * SEQ + s) * HDIM;
  float x1 = bf2f(src[srcRow + half]);
  float x2 = bf2f(src[srcRow + half + 64]);
  if (applyRope) {
    float invf = __powf(10000.0f, -(float)half * (1.0f / 64.0f));
    float ang  = (float)s * invf;
    float c = __cosf(ang), sn = __sinf(ang);
    dst[dstBase + half]      = f2bf(x1 * c - x2 * sn);
    dst[dstBase + half + 64] = f2bf(x1 * sn + x2 * c);
  } else {
    dst[dstBase + half]      = f2bf(x1);
    dst[dstBase + half + 64] = f2bf(x2);
  }
}

// ---------------------------------------------------------------------------
// Stage 4: causal flash attention. Grid (S/64, B*NH), block 128 (4 waves).
// Each wave owns 16 query rows; K/V tiles of 64 keys in LDS (V via async DMA);
// online softmax in fp32; both matmuls on WMMA.
// ---------------------------------------------------------------------------
__global__ __launch_bounds__(128)
void flash_attn(const bf16_t* __restrict__ Q, const bf16_t* __restrict__ Kr,
                const bf16_t* __restrict__ Vr, bf16_t* __restrict__ Out) {
  __shared__ bf16_t Kt[128 * 64];     // [d][key]  (transposed for B-fragments)
  __shared__ bf16_t Vt[64 * 128];     // [key][d]
  __shared__ bf16_t Pb[4][16 * 64];   // per-wave softmax tile

  const int lane = threadIdx.x & 31;
  const int w    = threadIdx.x >> 5;
  const int qblk = blockIdx.x;
  const int bh   = blockIdx.y;
  const size_t headOff = (size_t)bh * SEQ * HDIM;
  const int qbase = qblk * 64 + w * 16;
  const int kb8   = (lane >> 4) * 8;

  // Q fragments: 16 rows x 128 d = 4 K-chunks of 32, loaded straight from HBM
  v16bf aq[4];
  {
    const bf16_t* qp = Q + headOff + (size_t)(qbase + (lane & 15)) * HDIM;
#pragma unroll
    for (int kb = 0; kb < 4; ++kb)
      aq[kb] = cat16(*(const v8bf*)(qp + kb * 32 + kb8),
                     *(const v8bf*)(qp + kb * 32 + kb8 + 16));
  }

  float m_i[8], l_i[8];
  v8f o[8] = {};
#pragma unroll
  for (int r = 0; r < 8; ++r) { m_i[r] = -3.0e38f; l_i[r] = 0.0f; }

  const float scale = 0.08838834764831845f;  // 1/sqrt(128)
  const int rowBase = qbase + kb8;           // first of this lane's 8 rows

  for (int kt = 0; kt <= qblk; ++kt) {
    // ---- stage K (ds transpose) and V (async DMA) tiles ----
    {
      int key = threadIdx.x & 63;
      int dh  = (threadIdx.x >> 6) * 64;
      const bf16_t* kp = Kr + headOff + (size_t)(kt * 64 + key) * HDIM + dh;
      const bf16_t* vp = Vr + headOff + (size_t)(kt * 64 + key) * HDIM + dh;
#pragma unroll
      for (int c = 0; c < 4; ++c) {
        lds_copy16(vp + c * 16, &Vt[key * 128 + dh + c * 16]);
        union { uint4 u; bf16_t h[16]; } uk;
        uk.u = *(const uint4*)(kp + c * 16);
#pragma unroll
        for (int e = 0; e < 16; ++e) Kt[(dh + c * 16 + e) * 64 + key] = uk.h[e];
      }
    }
    async_wait0();
    __syncthreads();

    // ---- S = Q K^T  (16 queries x 64 keys, K-dim 128) ----
    v8f sAcc[4] = {};
#pragma unroll
    for (int kb = 0; kb < 4; ++kb) {
#pragma unroll
      for (int nb = 0; nb < 4; ++nb) {
        v16bf bk;
        for (int s = 0; s < 16; ++s)
          bk[s] = Kt[(kb * 32 + kb8 + (s & 7) + (s >> 3) * 16) * 64 + nb * 16 + (lane & 15)];
        sAcc[nb] = __builtin_amdgcn_wmma_f32_16x16x32_bf16(
            false, aq[kb], false, bk, (short)0, sAcc[nb], false, false);
      }
    }

    // ---- scale + causal mask ----
    const int colBase = kt * 64 + (lane & 15);
#pragma unroll
    for (int nb = 0; nb < 4; ++nb)
#pragma unroll
      for (int r = 0; r < 8; ++r) {
        float sv = sAcc[nb][r] * scale;
        if (colBase + nb * 16 > rowBase + r) sv = -1.0e30f;
        sAcc[nb][r] = sv;
      }

    // ---- online softmax + P staging ----
#pragma unroll
    for (int r = 0; r < 8; ++r) {
      float mx = fmaxf(fmaxf(sAcc[0][r], sAcc[1][r]),
                       fmaxf(sAcc[2][r], sAcc[3][r]));
      mx = half16_max(mx);
      float mnew  = fmaxf(m_i[r], mx);
      float alpha = __expf(m_i[r] - mnew);
      float rs = 0.0f;
#pragma unroll
      for (int nb = 0; nb < 4; ++nb) {
        float p = __expf(sAcc[nb][r] - mnew);
        sAcc[nb][r] = p;
        rs += p;
        Pb[w][(kb8 + r) * 64 + nb * 16 + (lane & 15)] = f2bf(p);
      }
      rs = half16_sum(rs);
      l_i[r] = l_i[r] * alpha + rs;
      m_i[r] = mnew;
#pragma unroll
      for (int j = 0; j < 8; ++j) o[j][r] *= alpha;
    }

    // ---- O += P V  (K-dim 64 keys, N-dim 128 d) ----
#pragma unroll
    for (int kb = 0; kb < 2; ++kb) {
      const bf16_t* pp = &Pb[w][(lane & 15) * 64 + kb * 32 + kb8];
      v16bf ap = cat16(*(const v8bf*)pp, *(const v8bf*)(pp + 16));
#pragma unroll
      for (int nb = 0; nb < 8; ++nb) {
        v16bf bv;
        for (int s = 0; s < 16; ++s)
          bv[s] = Vt[(kb * 32 + kb8 + (s & 7) + (s >> 3) * 16) * 128 + nb * 16 + (lane & 15)];
        o[nb] = __builtin_amdgcn_wmma_f32_16x16x32_bf16(
            false, ap, false, bv, (short)0, o[nb], false, false);
      }
    }
    __syncthreads();
  }

  // ---- normalize and write back to [B,S,NH*HD] layout (bf16) ----
  const int b = bh >> 4, h = bh & 15;
#pragma unroll
  for (int r = 0; r < 8; ++r) {
    int s = qbase + kb8 + r;
    float inv = 1.0f / l_i[r];
    size_t orow = ((size_t)b * SEQ + s) * HID + (size_t)h * HDIM;
#pragma unroll
    for (int nb = 0; nb < 8; ++nb)
      Out[orow + nb * 16 + (lane & 15)] = f2bf(o[nb][r] * inv);
  }
}

// ---------------------------------------------------------------------------
// Orchestration
// ---------------------------------------------------------------------------
extern "C" void kernel_launch(void* const* d_in, const int* in_sizes, int n_in,
                              void* d_out, int out_size, void* d_ws, size_t ws_size,
                              hipStream_t stream) {
  const float* hs = (const float*)d_in[0];
  const float* wq = (const float*)d_in[1];
  const float* wk = (const float*)d_in[2];
  const float* wv = (const float*)d_in[3];
  const float* wo = (const float*)d_in[4];

  char* ws = (char*)d_ws;
  const size_t SZ_X = (size_t)MTOT * HID * 2;  // 16 MB
  const size_t SZ_W = (size_t)HID * HID * 2;   // 8 MB

  bf16_t* Xb  = (bf16_t*)(ws);
  bf16_t* Wqb = (bf16_t*)(ws + SZ_X);
  bf16_t* Wkb = (bf16_t*)(ws + SZ_X + 1 * SZ_W);
  bf16_t* Wvb = (bf16_t*)(ws + SZ_X + 2 * SZ_W);
  bf16_t* Wob = (bf16_t*)(ws + SZ_X + 3 * SZ_W);
  char*  base = ws + SZ_X + 4 * SZ_W;
  bf16_t* Qb  = (bf16_t*)(base + 0 * SZ_X);
  bf16_t* Kb  = (bf16_t*)(base + 1 * SZ_X);
  bf16_t* Vb  = (bf16_t*)(base + 2 * SZ_X);
  bf16_t* Qr  = (bf16_t*)(base + 3 * SZ_X);
  bf16_t* Krp = (bf16_t*)(base + 4 * SZ_X);
  bf16_t* Vrp = (bf16_t*)(base + 5 * SZ_X);
  bf16_t* Ab  = (bf16_t*)(base + 6 * SZ_X);

  // Stage 1: fp32 -> bf16
  const int nH = MTOT * HID;  // 8,388,608
  const int nW = HID * HID;   // 4,194,304
  cvt_f32_bf16<<<nH / 4 / 256, 256, 0, stream>>>(hs, Xb, nH);
  cvt_f32_bf16<<<nW / 4 / 256, 256, 0, stream>>>(wq, Wqb, nW);
  cvt_f32_bf16<<<nW / 4 / 256, 256, 0, stream>>>(wk, Wkb, nW);
  cvt_f32_bf16<<<nW / 4 / 256, 256, 0, stream>>>(wv, Wvb, nW);
  cvt_f32_bf16<<<nW / 4 / 256, 256, 0, stream>>>(wo, Wob, nW);

  // Stage 2: Q/K/V projections
  dim3 gg(HID / 128, MTOT / 128);  // (16, 32)
  gemm_xwt<<<gg, 256, 0, stream>>>(Xb, Wqb, nullptr, Qb, MTOT, HID, HID);
  gemm_xwt<<<gg, 256, 0, stream>>>(Xb, Wkb, nullptr, Kb, MTOT, HID, HID);
  gemm_xwt<<<gg, 256, 0, stream>>>(Xb, Wvb, nullptr, Vb, MTOT, HID, HID);

  // Stage 3: RoPE + head-major transpose
  const int nPairs = BATCH * NHEAD * SEQ * 64;  // 4,194,304
  rope_transpose<<<nPairs / 256, 256, 0, stream>>>(Qb, Qr, 1);
  rope_transpose<<<nPairs / 256, 256, 0, stream>>>(Kb, Krp, 1);
  rope_transpose<<<nPairs / 256, 256, 0, stream>>>(Vb, Vrp, 0);

  // Stage 4: causal flash attention
  dim3 ga(SEQ / 64, BATCH * NHEAD);  // (32, 32)
  flash_attn<<<ga, 128, 0, stream>>>(Qr, Krp, Vrp, Ab);

  // Stage 5: output projection (fp32 out)
  gemm_xwt<<<gg, 256, 0, stream>>>(Ab, Wob, (float*)d_out, nullptr, MTOT, HID, HID);
}
